// EGNN_89412629168487
// MI455X (gfx1250) — matrix operand order repaired and checked by reading, not compiled, not run on hardware
//
#include <hip/hip_runtime.h>
#include <hip/hip_bf16.h>
#include <math.h>

typedef __bf16 bf16_t;
typedef __attribute__((ext_vector_type(16))) __bf16 v16bf;
typedef __attribute__((ext_vector_type(8)))  float  v8f;

#define NN   10000
#define EE   80000
#define GG   64
#define HID  256
#define DEPTH 6
#define KPAD1 800   // 784 padded to 25*32

__device__ __forceinline__ float silu_f(float x) { return x / (1.f + __expf(-x)); }

// Store value for logical (row, k) of an M x K matrix into LDS laid out as per-lane
// WMMA 16-bit A fragments: [rtile][kstep][lane][16 halves], lane = m + 16*((k%32)>>4),
// half j = k&15  (inverse of K = hi*16 + j; this file's contiguous k order, matched by B).
__device__ __forceinline__ void storeA(bf16_t* A, int ksteps, int row, int k, float v) {
    int rtile = row >> 4, m = row & 15;
    int ks = k >> 5, kl = k & 31;
    int j  = kl & 15;
    int ln = m + ((kl >> 4) << 4);
    A[(((size_t)(rtile * ksteps + ks) * 32) + ln) * 16 + j] = (bf16_t)v;
}

__device__ __forceinline__ void zero_acc(v8f* acc) {
#pragma unroll
    for (int c = 0; c < 4; ++c)
#pragma unroll
        for (int i = 0; i < 8; ++i) acc[c][i] = 0.f;
}

// Wave computes one 16-row tile x four 16-col tiles. A from LDS (pre-swizzled),
// B from global bf16 weights stored transposed [n][kpad] (contiguous per lane).
__device__ __forceinline__ void gemm4(const bf16_t* __restrict__ Wt, const bf16_t* aA,
                                      int ksteps, int rtile, int ctBase, int lane, v8f* acc) {
    const int kpad = ksteps * 32;
    const int nl = lane & 15, kgrp = (lane >> 4) * 16;
    for (int ks = 0; ks < ksteps; ++ks) {
        v16bf a = *(const v16bf*)(aA + (((size_t)(rtile * ksteps + ks) * 32) + lane) * 16);
        int krow = ks * 32 + kgrp;
#pragma unroll
        for (int ct = 0; ct < 4; ++ct) {
            int n = (ctBase + ct) * 16 + nl;
            v16bf b = *(const v16bf*)(Wt + (size_t)n * kpad + krow);
            acc[ct] = __builtin_amdgcn_wmma_f32_16x16x32_bf16(false, a, false, b,
                                                              (short)0, acc[ct], false, false);
        }
    }
}

// silu(acc + bias) -> LDS in A layout (feeds the next GEMM)
__device__ __forceinline__ void epiloA(const v8f* acc, const float* bias, bf16_t* dstA,
                                       int ks2, int rtile, int ctBase, int lane) {
    int hib = lane >> 4, nl = lane & 15;
#pragma unroll
    for (int ct = 0; ct < 4; ++ct) {
        int n = (ctBase + ct) * 16 + nl;
        float b = bias[n];
#pragma unroll
        for (int i = 0; i < 8; ++i) {
            int m = hib * 8 + i;  // row within rtile
            storeA(dstA + (size_t)0, ks2, rtile * 16 + m, n, silu_f(acc[ct][i] + b));
        }
    }
}

// silu(acc + bias) -> f32 LDS [32][256]
__device__ __forceinline__ void epiloZ(const v8f* acc, const float* bias, float* zbuf,
                                       int rtile, int ctBase, int lane) {
    int hib = lane >> 4, nl = lane & 15;
#pragma unroll
    for (int ct = 0; ct < 4; ++ct) {
        int n = (ctBase + ct) * 16 + nl;
        float b = bias[n];
#pragma unroll
        for (int i = 0; i < 8; ++i) {
            int m = rtile * 16 + hib * 8 + i;
            zbuf[m * 256 + n] = silu_f(acc[ct][i] + b);
        }
    }
}

// ---------------- preprocessing kernels ----------------

// convert (D,K,N) f32 weights to bf16 transposed (D,N,Kpad), zero-padded in K
__global__ void k_wt(const float* __restrict__ src, bf16_t* __restrict__ dst,
                     int K, int Nc, int Kpad, int total) {
    int idx = blockIdx.x * 256 + threadIdx.x;
    if (idx >= total) return;
    int nk = Nc * Kpad;
    int d = idx / nk, r = idx - d * nk;
    int n = r / Kpad, k = r - n * Kpad;
    float v = (k < K) ? src[((size_t)d * K + k) * Nc + n] : 0.f;
    dst[idx] = (bf16_t)v;
}

// A1 = atom_W @ fuse_W[0:85], A2 = charge_W @ fuse_W[85:170], B1e = bond_W @ lift_W
__global__ void k_tables(const float* atom_W, const float* charge_W, const float* bond_W,
                         const float* fuse_W, const float* lift_W,
                         float* A1, float* A2, float* B1e) {
    int j = threadIdx.x;  // 0..255
    for (int r = 0; r < 16; ++r) {
        float s = 0.f;
        for (int t = 0; t < 85; ++t) s += atom_W[r * 85 + t] * fuse_W[t * 256 + j];
        A1[r * 256 + j] = s;
    }
    for (int r = 0; r < 6; ++r) {
        float s = 0.f;
        for (int t = 0; t < 85; ++t) s += charge_W[r * 85 + t] * fuse_W[(85 + t) * 256 + j];
        A2[r * 256 + j] = s;
    }
    for (int r = 0; r < 5; ++r) {
        float s = 0.f;
        for (int t = 0; t < 85; ++t) s += bond_W[r * 85 + t] * lift_W[t * 256 + j];
        B1e[r * 256 + j] = s;
    }
}

// gumbel-hard argmax for nodes + copy x
__global__ void k_node_meta(const float* a_t, const float* c_t, const float* g_a, const float* g_c,
                            const float* x_t, int* iA, int* iC, float* x, int n) {
    int i = blockIdx.x * 256 + threadIdx.x;
    if (i >= n) return;
    float best = -1e30f; int bi = 0;
    for (int k = 0; k < 16; ++k) {
        float v = __logf(fmaxf(a_t[i * 16 + k], 1e-12f)) + g_a[i * 16 + k];
        if (v > best) { best = v; bi = k; }
    }
    iA[i] = bi;
    best = -1e30f; bi = 0;
    for (int k = 0; k < 6; ++k) {
        float v = __logf(fmaxf(c_t[i * 6 + k], 1e-12f)) + g_c[i * 6 + k];
        if (v > best) { best = v; bi = k; }
    }
    iC[i] = bi;
    x[i * 3 + 0] = x_t[i * 3 + 0];
    x[i * 3 + 1] = x_t[i * 3 + 1];
    x[i * 3 + 2] = x_t[i * 3 + 2];
}

__global__ void k_edge_meta(const float* e_t, const float* g_e, int* iE, int n) {
    int i = blockIdx.x * 256 + threadIdx.x;
    if (i >= n) return;
    float best = -1e30f; int bi = 0;
    for (int k = 0; k < 5; ++k) {
        float v = __logf(fmaxf(e_t[i * 5 + k], 1e-12f)) + g_e[i * 5 + k];
        if (v > best) { best = v; bi = k; }
    }
    iE[i] = bi;
}

__global__ void k_encode_h(const int* iA, const int* iC, const float* A1, const float* A2,
                           const float* fuse_b, float* h) {
    int n = blockIdx.x, j = threadIdx.x;
    h[(size_t)n * 256 + j] = A1[iA[n] * 256 + j] + A2[iC[n] * 256 + j] + fuse_b[j];
}

__global__ void k_zero(float* p, int n) {
    int i = blockIdx.x * 256 + threadIdx.x;
    if (i < n) p[i] = 0.f;
}

// ---------------- per-depth edge kernel ----------------

struct __align__(64) EdgeSmem {
    bf16_t catA[2 * 25 * 32 * 16];  // 32x800 tile in A layout (50 KB)
    bf16_t midA[2 * 8 * 32 * 16];   // 32x256 (16 KB)
    bf16_t mbuf[2 * 8 * 32 * 16];   // 32x256 (16 KB)
    float  zbuf[32 * 256];          // 32 KB
    int    srcI[32], dstI[32], ieI[32];
    float  rijL[32 * 3], distL[32];
};

__global__ __launch_bounds__(256) void k_edge(
    int d, const int* __restrict__ src, const int* __restrict__ dst, const int* __restrict__ iE,
    const float* __restrict__ h, const float* __restrict__ x,
    const float* __restrict__ B1e, const float* __restrict__ lift_b,
    const bf16_t* __restrict__ phimW1t, const bf16_t* __restrict__ phimW2t,
    const bf16_t* __restrict__ phixW1t,
    const bf16_t* __restrict__ psimW1t, const bf16_t* __restrict__ psimW2t,
    const float* __restrict__ phim_b1, const float* __restrict__ phim_b2,
    const float* __restrict__ phix_b1, const float* __restrict__ phix_W2,
    const float* __restrict__ phix_b2,
    const float* __restrict__ psim_b1, const float* __restrict__ psim_b2,
    float* __restrict__ pm_sum, float* __restrict__ dx_sum) {
    __shared__ EdgeSmem s;
    int t = threadIdx.x, lane = t & 31, w = t >> 5;
    int rtile = w & 1, ctBase = (w >> 1) * 4;
    int e0 = blockIdx.x * 32;

    if (t < 32) {
        int e = e0 + t;
        int si = src[e], di = dst[e];
        s.srcI[t] = si; s.dstI[t] = di; s.ieI[t] = iE[e];
        float rx = x[di * 3 + 0] - x[si * 3 + 0];
        float ry = x[di * 3 + 1] - x[si * 3 + 1];
        float rz = x[di * 3 + 2] - x[si * 3 + 2];
        s.rijL[t * 3 + 0] = rx; s.rijL[t * 3 + 1] = ry; s.rijL[t * 3 + 2] = rz;
        s.distL[t] = sqrtf(fmaxf(rx * rx + ry * ry + rz * rz, 0.f));
    }
    __syncthreads();

    // build cat = [h[dst], h[src], rbf, e_emb] (+pad) directly in A layout
    for (int idx = t; idx < 32 * KPAD1; idx += 256) {
        int row = idx / KPAD1, k = idx - row * KPAD1;
        float v;
        if (k < 256)       v = h[(size_t)s.dstI[row] * 256 + k];
        else if (k < 512)  v = h[(size_t)s.srcI[row] * 256 + (k - 256)];
        else if (k < 528) {
            int kk = k - 512;
            float c = kk * (10.f / 15.f);
            float z = (s.distL[row] - c) / (0.625f + 1e-12f);
            v = __expf(-0.5f * z * z);
        } else if (k < 784) {
            int j2 = k - 528;
            v = B1e[s.ieI[row] * 256 + j2] + lift_b[j2];
        } else v = 0.f;
        storeA(s.catA, 25, row, k, v);
    }
    __syncthreads();

    v8f acc[4];
    // m = silu(silu(cat@W1+b1)@W2+b2)
    zero_acc(acc);
    gemm4(phimW1t + (size_t)d * 256 * KPAD1, s.catA, 25, rtile, ctBase, lane, acc);
    epiloA(acc, phim_b1 + d * 256, s.midA, 8, rtile, ctBase, lane);
    __syncthreads();
    zero_acc(acc);
    gemm4(phimW2t + (size_t)d * 256 * 256, s.midA, 8, rtile, ctBase, lane, acc);
    epiloA(acc, phim_b2 + d * 256, s.mbuf, 8, rtile, ctBase, lane);
    __syncthreads();
    // gate pre-act: silu(m@phix_W1 + bx1) -> zbuf
    zero_acc(acc);
    gemm4(phixW1t + (size_t)d * 256 * 256, s.mbuf, 8, rtile, ctBase, lane, acc);
    epiloZ(acc, phix_b1 + d * 256, s.zbuf, rtile, ctBase, lane);
    __syncthreads();
    if (t < 32) {  // gate scalar + dx scatter
        const float* w2 = phix_W2 + d * 256;
        float sdot = 0.f;
        for (int n = 0; n < 256; ++n) sdot += s.zbuf[t * 256 + n] * w2[n];
        float gate = sdot + phix_b2[d];
        int di = s.dstI[t];
        for (int c = 0; c < 3; ++c)
            unsafeAtomicAdd(&dx_sum[di * 3 + c], s.rijL[t * 3 + c] * gate);
    }
    // pm = silu(silu(cat@psiW1+c1)@psiW2+c2)
    zero_acc(acc);
    gemm4(psimW1t + (size_t)d * 256 * KPAD1, s.catA, 25, rtile, ctBase, lane, acc);
    epiloA(acc, psim_b1 + d * 256, s.midA, 8, rtile, ctBase, lane);
    __syncthreads();
    zero_acc(acc);
    gemm4(psimW2t + (size_t)d * 256 * 256, s.midA, 8, rtile, ctBase, lane, acc);
    {
        const float* b2 = psim_b2 + d * 256;
        int hib = lane >> 4, nl = lane & 15;
#pragma unroll
        for (int ct = 0; ct < 4; ++ct) {
            int n = (ctBase + ct) * 16 + nl;
            float b = b2[n];
#pragma unroll
            for (int i = 0; i < 8; ++i) {
                int m = rtile * 16 + hib * 8 + i;
                unsafeAtomicAdd(&pm_sum[(size_t)s.dstI[m] * 256 + n], silu_f(acc[ct][i] + b));
            }
        }
    }
}

// ---------------- per-depth node kernel ----------------

struct __align__(64) NodeSmem {
    bf16_t catA[2 * 16 * 32 * 16];  // 32x512 (32 KB)
    bf16_t midA[2 * 8 * 32 * 16];   // 16 KB
    float  zbuf[32 * 256];          // 32 KB
    float  red[32][8];
    float  muL[32], invL[32];
};

__global__ __launch_bounds__(256) void k_node(
    int d, int Nn, float* __restrict__ h, float* __restrict__ x,
    const float* __restrict__ pm_sum, const float* __restrict__ dx_sum,
    const bf16_t* __restrict__ updhW1t, const bf16_t* __restrict__ updhW2t,
    const float* __restrict__ updh_b1, const float* __restrict__ updh_b2,
    const float* __restrict__ ln_g, const float* __restrict__ ln_b) {
    __shared__ NodeSmem s;
    int t = threadIdx.x, lane = t & 31, w = t >> 5;
    int rtile = w & 1, ctBase = (w >> 1) * 4;
    int n0 = blockIdx.x * 32;

    for (int idx = t; idx < 32 * 512; idx += 256) {
        int row = idx >> 9, k = idx & 511;
        int nid = n0 + row;
        float v = 0.f;
        if (nid < Nn) v = (k < 256) ? h[(size_t)nid * 256 + k] : pm_sum[(size_t)nid * 256 + (k - 256)];
        storeA(s.catA, 16, row, k, v);
    }
    __syncthreads();

    v8f acc[4];
    zero_acc(acc);
    gemm4(updhW1t + (size_t)d * 256 * 512, s.catA, 16, rtile, ctBase, lane, acc);
    epiloA(acc, updh_b1 + d * 256, s.midA, 8, rtile, ctBase, lane);
    __syncthreads();
    zero_acc(acc);
    gemm4(updhW2t + (size_t)d * 256 * 256, s.midA, 8, rtile, ctBase, lane, acc);
    {   // z2 = h + u
        const float* b2 = updh_b2 + d * 256;
        int hib = lane >> 4, nl = lane & 15;
#pragma unroll
        for (int ct = 0; ct < 4; ++ct) {
            int n = (ctBase + ct) * 16 + nl;
            float b = b2[n];
#pragma unroll
            for (int i = 0; i < 8; ++i) {
                int m = rtile * 16 + hib * 8 + i;
                int nid = n0 + m;
                float hv = (nid < Nn) ? h[(size_t)nid * 256 + n] : 0.f;
                s.zbuf[m * 256 + n] = hv + acc[ct][i] + b;
            }
        }
    }
    __syncthreads();
    // LayerNorm over 256 features
    { int row = t >> 3, sub = t & 7; float sm = 0.f;
      for (int c = sub * 32; c < sub * 32 + 32; ++c) sm += s.zbuf[row * 256 + c];
      s.red[row][sub] = sm; }
    __syncthreads();
    if (t < 32) { float sm = 0.f; for (int j = 0; j < 8; ++j) sm += s.red[t][j];
                  s.muL[t] = sm * (1.f / 256.f); }
    __syncthreads();
    { int row = t >> 3, sub = t & 7; float mu = s.muL[row]; float sm = 0.f;
      for (int c = sub * 32; c < sub * 32 + 32; ++c) { float dd = s.zbuf[row * 256 + c] - mu; sm += dd * dd; }
      s.red[row][sub] = sm; }
    __syncthreads();
    if (t < 32) { float sm = 0.f; for (int j = 0; j < 8; ++j) sm += s.red[t][j];
                  s.invL[t] = rsqrtf(sm * (1.f / 256.f) + 1e-5f); }
    __syncthreads();
    for (int idx = t; idx < 32 * 256; idx += 256) {
        int row = idx >> 8, n = idx & 255;
        int nid = n0 + row;
        if (nid < Nn)
            h[(size_t)nid * 256 + n] =
                (s.zbuf[idx] - s.muL[row]) * s.invL[row] * ln_g[d * 256 + n] + ln_b[d * 256 + n];
    }
    if (t < 96) {
        int row = t / 3, c = t - row * 3;
        int nid = n0 + row;
        if (nid < Nn) x[nid * 3 + c] += dx_sum[nid * 3 + c];
    }
}

// ---------------- readout ----------------

__global__ void k_accum(const float* __restrict__ h, const int* __restrict__ ng,
                        float* __restrict__ hg, float* __restrict__ cnt, int Nn) {
    int n = blockIdx.x, t = threadIdx.x;
    int g = ng[n];
    unsafeAtomicAdd(&hg[(size_t)g * 256 + t], h[(size_t)n * 256 + t]);
    if (t == 0) unsafeAtomicAdd(&cnt[g], 1.f);
}

__global__ void k_head(const float* __restrict__ hg, const float* __restrict__ cnt,
                       const float* __restrict__ head_W, const float* __restrict__ head_b,
                       float* __restrict__ out) {
    __shared__ float red[256];
    int g = blockIdx.x, t = threadIdx.x;
    float c = fmaxf(cnt[g], 1.f);
    red[t] = (hg[(size_t)g * 256 + t] / c) * head_W[t];
    __syncthreads();
    for (int sft = 128; sft > 0; sft >>= 1) {
        if (t < sft) red[t] += red[t + sft];
        __syncthreads();
    }
    if (t == 0) {
        float v = (red[0] + head_b[0]) * 0.5f;
        out[g] = 1.f / (1.f + __expf(-v));
    }
}

// ---------------- host ----------------

extern "C" void kernel_launch(void* const* d_in, const int* in_sizes, int n_in,
                              void* d_out, int out_size, void* d_ws, size_t ws_size,
                              hipStream_t stream) {
    (void)in_sizes; (void)n_in; (void)out_size; (void)ws_size;
    const float* a_t     = (const float*)d_in[0];
    const float* c_t     = (const float*)d_in[1];
    const float* e_t     = (const float*)d_in[2];
    const float* x_t     = (const float*)d_in[3];
    const float* g_a     = (const float*)d_in[4];
    const float* g_c     = (const float*)d_in[5];
    const float* g_e     = (const float*)d_in[6];
    const float* atom_W  = (const float*)d_in[7];
    const float* charge_W= (const float*)d_in[8];
    const float* bond_W  = (const float*)d_in[9];
    const float* fuse_W  = (const float*)d_in[10];
    const float* fuse_b  = (const float*)d_in[11];
    const float* lift_W  = (const float*)d_in[12];
    const float* lift_b  = (const float*)d_in[13];
    const float* phim_W1 = (const float*)d_in[14];
    const float* phim_b1 = (const float*)d_in[15];
    const float* phim_W2 = (const float*)d_in[16];
    const float* phim_b2 = (const float*)d_in[17];
    const float* phix_W1 = (const float*)d_in[18];
    const float* phix_b1 = (const float*)d_in[19];
    const float* phix_W2 = (const float*)d_in[20];
    const float* phix_b2 = (const float*)d_in[21];
    const float* psim_W1 = (const float*)d_in[22];
    const float* psim_b1 = (const float*)d_in[23];
    const float* psim_W2 = (const float*)d_in[24];
    const float* psim_b2 = (const float*)d_in[25];
    const float* updh_W1 = (const float*)d_in[26];
    const float* updh_b1 = (const float*)d_in[27];
    const float* updh_W2 = (const float*)d_in[28];
    const float* updh_b2 = (const float*)d_in[29];
    const float* ln_g    = (const float*)d_in[30];
    const float* ln_b    = (const float*)d_in[31];
    const float* head_W  = (const float*)d_in[32];
    const float* head_b  = (const float*)d_in[33];
    const int*   src     = (const int*)d_in[34];
    const int*   dst     = (const int*)d_in[35];
    const int*   ng      = (const int*)d_in[36];
    float* out = (float*)d_out;

    char* ws = (char*)d_ws;
    size_t off = 0;
    auto carve = [&](size_t bytes) { char* p = ws + off; off = (off + bytes + 255) & ~(size_t)255; return p; };

    float* h      = (float*)carve((size_t)NN * 256 * 4);
    float* xw     = (float*)carve((size_t)NN * 3 * 4);
    float* pm_sum = (float*)carve((size_t)NN * 256 * 4);
    float* dx_sum = (float*)carve((size_t)NN * 3 * 4);
    float* A1     = (float*)carve(16 * 256 * 4);
    float* A2     = (float*)carve(6 * 256 * 4);
    float* B1e    = (float*)carve(5 * 256 * 4);
    int*   iA     = (int*)carve((size_t)NN * 4);
    int*   iC     = (int*)carve((size_t)NN * 4);
    int*   iEd    = (int*)carve((size_t)EE * 4);
    float* hg     = (float*)carve((size_t)GG * 256 * 4);
    float* cnt    = (float*)carve((size_t)GG * 4);
    bf16_t* phimW1t = (bf16_t*)carve((size_t)DEPTH * 256 * KPAD1 * 2);
    bf16_t* psimW1t = (bf16_t*)carve((size_t)DEPTH * 256 * KPAD1 * 2);
    bf16_t* phimW2t = (bf16_t*)carve((size_t)DEPTH * 256 * 256 * 2);
    bf16_t* psimW2t = (bf16_t*)carve((size_t)DEPTH * 256 * 256 * 2);
    bf16_t* phixW1t = (bf16_t*)carve((size_t)DEPTH * 256 * 256 * 2);
    bf16_t* updhW1t = (bf16_t*)carve((size_t)DEPTH * 256 * 512 * 2);
    bf16_t* updhW2t = (bf16_t*)carve((size_t)DEPTH * 256 * 256 * 2);

    // weight conversion (bf16, transposed, K-padded)
    int tot1 = DEPTH * 256 * KPAD1;
    int tot2 = DEPTH * 256 * 256;
    int tot3 = DEPTH * 256 * 512;
    k_wt<<<(tot1 + 255) / 256, 256, 0, stream>>>(phim_W1, phimW1t, 784, 256, KPAD1, tot1);
    k_wt<<<(tot1 + 255) / 256, 256, 0, stream>>>(psim_W1, psimW1t, 784, 256, KPAD1, tot1);
    k_wt<<<(tot2 + 255) / 256, 256, 0, stream>>>(phim_W2, phimW2t, 256, 256, 256, tot2);
    k_wt<<<(tot2 + 255) / 256, 256, 0, stream>>>(psim_W2, psimW2t, 256, 256, 256, tot2);
    k_wt<<<(tot2 + 255) / 256, 256, 0, stream>>>(phix_W1, phixW1t, 256, 256, 256, tot2);
    k_wt<<<(tot3 + 255) / 256, 256, 0, stream>>>(updh_W1, updhW1t, 512, 256, 512, tot3);
    k_wt<<<(tot2 + 255) / 256, 256, 0, stream>>>(updh_W2, updhW2t, 256, 256, 256, tot2);

    // encoder (gumbel-hard == row lookup in forward)
    k_tables<<<1, 256, 0, stream>>>(atom_W, charge_W, bond_W, fuse_W, lift_W, A1, A2, B1e);
    k_node_meta<<<(NN + 255) / 256, 256, 0, stream>>>(a_t, c_t, g_a, g_c, x_t, iA, iC, xw, NN);
    k_edge_meta<<<(EE + 255) / 256, 256, 0, stream>>>(e_t, g_e, iEd, EE);
    k_encode_h<<<NN, 256, 0, stream>>>(iA, iC, A1, A2, fuse_b, h);

    for (int d = 0; d < DEPTH; ++d) {
        k_zero<<<(NN * 256 + 255) / 256, 256, 0, stream>>>(pm_sum, NN * 256);
        k_zero<<<(NN * 3 + 255) / 256, 256, 0, stream>>>(dx_sum, NN * 3);
        k_edge<<<EE / 32, 256, 0, stream>>>(d, src, dst, iEd, h, xw, B1e, lift_b,
                                            phimW1t, phimW2t, phixW1t, psimW1t, psimW2t,
                                            phim_b1, phim_b2, phix_b1, phix_W2, phix_b2,
                                            psim_b1, psim_b2, pm_sum, dx_sum);
        k_node<<<(NN + 31) / 32, 256, 0, stream>>>(d, NN, h, xw, pm_sum, dx_sum,
                                                   updhW1t, updhW2t, updh_b1, updh_b2, ln_g, ln_b);
    }

    k_zero<<<(GG * 256 + 255) / 256, 256, 0, stream>>>(hg, GG * 256);
    k_zero<<<1, 256, 0, stream>>>(cnt, GG);
    k_accum<<<NN, 256, 0, stream>>>(h, ng, hg, cnt, NN);
    k_head<<<GG, 256, 0, stream>>>(hg, cnt, head_W, head_b, out);
}
